// LIFLayerTrain_27625229648272
// MI455X (gfx1250) — matrix operand order repaired and checked by reading, not compiled, not run on hardware
//
#include <hip/hip_runtime.h>
#include <hip/hip_bf16.h>

// ---------------- types ----------------
typedef __bf16 bf16;
typedef __attribute__((ext_vector_type(16))) bf16  bf16x16;
typedef __attribute__((ext_vector_type(4)))  float f32x4;
typedef __attribute__((ext_vector_type(8)))  float f32x8;
typedef __attribute__((ext_vector_type(4)))  unsigned int u32x4;
typedef int v4i __attribute__((vector_size(16)));   // matches builtin's int4 param

union BF16x16 { bf16x16 v; f32x4 q[2]; };

// ---------------- problem constants ----------------
#define BATCH   256
#define TSTEPS  100
#define IDIM    2048
#define HDIM    2048

#define M_ROWS  112          // 7 * 16 (covers T=100, rows 100..111 are junk, never read)
#define M_TILES 7
#define KSTEP   32
#define KITERS  (IDIM / KSTEP)   // 64
#define TILE_N  128          // H columns per workgroup (8 waves x 16 cols)
#define ALD     40           // LDS A row stride in bf16 (padded: conflict-free b128 reads)
#define PLD     132          // LDS proj row stride in f32 (padded)

#define A_BUF_ELEMS (M_ROWS * ALD)                            // per buffer
#define SMEM_BYTES  (2 * A_BUF_ELEMS * 2 + M_ROWS * PLD * 4)  // 17920 + 59136 = 77056

#define THRESHOLD 1.0f
#define DECAY     0.9f

// ---------------- CDNA5 async global->LDS path (guarded) ----------------
#if defined(__HIP_DEVICE_COMPILE__) && \
    __has_builtin(__builtin_amdgcn_global_load_async_to_lds_b128)
  #define HAVE_ASYNC_LDS 1
  #define AS1_V4(p) ((__attribute__((address_space(1))) v4i*)(uintptr_t)(p))
  #define AS3_V4(p) ((__attribute__((address_space(3))) v4i*)(p))
  #if __has_builtin(__builtin_amdgcn_s_wait_asynccnt)
    #define WAIT_ASYNC0() __builtin_amdgcn_s_wait_asynccnt(0)
  #else
    #define WAIT_ASYNC0() asm volatile("s_wait_asynccnt 0x0" ::: "memory")
  #endif
#endif

// ---------------- f32 -> bf16 (RNE) convert kernel ----------------
__device__ __forceinline__ unsigned int bf16_rne_bits(float f) {
  unsigned int u = __float_as_uint(f);
  return (u + 0x7FFFu + ((u >> 16) & 1u)) >> 16;
}

__global__ void __launch_bounds__(256)
cvt_f32_to_bf16(const float* __restrict__ src, unsigned int* __restrict__ dst, long n) {
  long i = ((long)blockIdx.x * blockDim.x + threadIdx.x);
  long base = i * 8;
  if (base >= n) return;
  const f32x4* s = (const f32x4*)(src + base);
  f32x4 x0 = s[0];
  f32x4 x1 = s[1];
  u32x4 o;
  o.x = bf16_rne_bits(x0.x) | (bf16_rne_bits(x0.y) << 16);
  o.y = bf16_rne_bits(x0.z) | (bf16_rne_bits(x0.w) << 16);
  o.z = bf16_rne_bits(x1.x) | (bf16_rne_bits(x1.y) << 16);
  o.w = bf16_rne_bits(x1.z) | (bf16_rne_bits(x1.w) << 16);
  *(u32x4*)(dst + i * 4) = o;
}

// ---------------- fused GEMM (WMMA bf16) + LIF scan ----------------
// Grid: (HDIM/TILE_N, BATCH), block: 256 threads (8 waves).
// Wave w owns N-tile w (16 cols) and all 7 M-tiles.
__global__ void __launch_bounds__(256)
lif_fused(const bf16* __restrict__ Abf,   // [BATCH*TSTEPS, IDIM] bf16
          const bf16* __restrict__ Wbf,   // [HDIM, IDIM] bf16
          const float* __restrict__ bias, // [HDIM]
          float* __restrict__ spikes,     // [BATCH, TSTEPS, HDIM]
          float* __restrict__ hfinal) {   // [BATCH, HDIM]
  extern __shared__ unsigned char smem_raw[];
  bf16*  AL   = (bf16*)smem_raw;                                     // [2][M_ROWS*ALD]
  float* PROJ = (float*)(smem_raw + 2 * A_BUF_ELEMS * sizeof(bf16)); // [M_ROWS*PLD]

  const int tid   = threadIdx.x;
  const int wave  = tid >> 5;
  const int lane  = tid & 31;
  const int b     = blockIdx.y;
  const int hbase = blockIdx.x * TILE_N;

  // ---- cooperative A staging: 2 threads per row, 32B (16 bf16) each ----
  const int  r      = tid >> 1;          // row 0..127 (active if < M_ROWS)
  const int  half   = tid & 1;
  const bool stager = (r < M_ROWS);
  const bf16* gA = Abf + (size_t)(b * TSTEPS + r) * IDIM + half * 16;

  // ---- B fragment addressing (direct global, K-contiguous per lane) ----
  const int   bn = lane & 15;
  const int   bk = (lane >> 4) * 16;
  const bf16* gB = Wbf + (size_t)(hbase + wave * 16 + bn) * IDIM + bk;

  // ---- A fragment addressing in LDS ----
  const int am = lane & 15;
  const int ac = (lane >> 4) * 16;       // K chunk base within k-step

  f32x8 acc[M_TILES];
#pragma unroll
  for (int mt = 0; mt < M_TILES; ++mt) acc[mt] = (f32x8)(0.0f);

  // ---- prologue: stage k-step 0 and prefetch B fragment for k-step 0 ----
#ifdef HAVE_ASYNC_LDS
  if (stager) {
    bf16* l = AL + r * ALD + half * 16;
    __builtin_amdgcn_global_load_async_to_lds_b128(AS1_V4(gA), AS3_V4(l), 0, 0);
    __builtin_amdgcn_global_load_async_to_lds_b128(AS1_V4(gA), AS3_V4(l), 16, 0);
  }
#else
  f32x4 p0, p1;
  if (stager) {
    p0 = *(const f32x4*)(gA + 0);
    p1 = *(const f32x4*)(gA + 8);
  }
#endif
  f32x4 bq0 = *(const f32x4*)(gB + 0);
  f32x4 bq1 = *(const f32x4*)(gB + 8);

  int buf = 0;
  for (int ks = 0; ks < KITERS; ++ks) {
#ifdef HAVE_ASYNC_LDS
    // data for AL[buf] was issued async (prologue or previous iteration)
    WAIT_ASYNC0();
    __syncthreads();
    // issue async staging for next k-step into the other buffer
    if (ks + 1 < KITERS && stager) {
      const bf16* g = gA + (size_t)(ks + 1) * KSTEP;
      bf16*       l = AL + (buf ^ 1) * A_BUF_ELEMS + r * ALD + half * 16;
      __builtin_amdgcn_global_load_async_to_lds_b128(AS1_V4(g), AS3_V4(l), 0, 0);
      __builtin_amdgcn_global_load_async_to_lds_b128(AS1_V4(g), AS3_V4(l), 16, 0);
    }
#else
    // stage prefetched A chunk into LDS buffer `buf`
    if (stager) {
      bf16* w = AL + buf * A_BUF_ELEMS + r * ALD + half * 16;
      *(f32x4*)(w + 0) = p0;
      *(f32x4*)(w + 8) = p1;
    }
    __syncthreads();
    if (ks + 1 < KITERS && stager) {
      const bf16* g = gA + (size_t)(ks + 1) * KSTEP;
      p0 = *(const f32x4*)(g + 0);
      p1 = *(const f32x4*)(g + 8);
    }
#endif

    // B fragment for this k-step was prefetched one iteration ago;
    // issue next-iteration B prefetch early.
    BF16x16 bfr;
    bfr.q[0] = bq0;
    bfr.q[1] = bq1;
    if (ks + 1 < KITERS) {
      const bf16* g = gB + (size_t)(ks + 1) * KSTEP;
      bq0 = *(const f32x4*)(g + 0);
      bq1 = *(const f32x4*)(g + 8);
    }

    // Load ALL seven A fragments first (distinct registers), then issue the
    // WMMAs: lets the scheduler stagger s_wait_dscnt instead of draining to 0
    // before every WMMA.
    const bf16* ALb = AL + buf * A_BUF_ELEMS;
    BF16x16 afr[M_TILES];
#pragma unroll
    for (int mt = 0; mt < M_TILES; ++mt) {
      const bf16* ap = ALb + (mt * 16 + am) * ALD + ac;
      afr[mt].q[0] = *(const f32x4*)(ap + 0);    // K = c .. c+7
      afr[mt].q[1] = *(const f32x4*)(ap + 8);    // K = c+8 .. c+15
    }
#pragma unroll
    for (int mt = 0; mt < M_TILES; ++mt) {
      acc[mt] = __builtin_amdgcn_wmma_f32_16x16x32_bf16(
          false, afr[mt].v, false, bfr.v, (short)0, acc[mt], false, false);
    }
    buf ^= 1;
  }

  // ---- bias add + spill proj tile to LDS ----
  const int cn     = lane & 15;
  const int col    = wave * 16 + cn;
  const int rowoff = (lane >> 4) * 8;
  const float bv   = bias[hbase + col];
#pragma unroll
  for (int mt = 0; mt < M_TILES; ++mt) {
#pragma unroll
    for (int j = 0; j < 8; ++j) {
      int m = mt * 16 + rowoff + j;
      PROJ[m * PLD + col] = acc[mt][j] + bv;
    }
  }
  __syncthreads();

  // ---- sequential LIF scan out of LDS: one thread per H column ----
  if (tid < TILE_N) {
    const int hg = hbase + tid;
    float h = 0.0f;
    float* sp = spikes + (size_t)b * TSTEPS * HDIM + hg;
    for (int t = 0; t < TSTEPS; ++t) {
      float p = PROJ[t * PLD + tid];
      h = DECAY * h + p;
      float s = (h >= THRESHOLD) ? 1.0f : 0.0f;
      h *= (1.0f - s);
      sp[(size_t)t * HDIM] = s;
    }
    hfinal[(size_t)b * HDIM + hg] = h;
  }
}

// ---------------- host launch ----------------
extern "C" void kernel_launch(void* const* d_in, const int* in_sizes, int n_in,
                              void* d_out, int out_size, void* d_ws, size_t ws_size,
                              hipStream_t stream) {
  const float* inp  = (const float*)d_in[0];  // [256,100,2048]
  const float* W    = (const float*)d_in[1];  // [2048,2048]
  const float* bias = (const float*)d_in[2];  // [2048]

  float* spikes = (float*)d_out;
  float* hfinal = spikes + (size_t)BATCH * TSTEPS * HDIM;

  // workspace: bf16 inputs then bf16 W (~108 MB)
  bf16* Abf = (bf16*)d_ws;
  bf16* Wbf = Abf + (size_t)BATCH * TSTEPS * IDIM;

  long nA = (long)in_sizes[0];   // 52,428,800
  long nW = (long)in_sizes[1];   //  4,194,304

  cvt_f32_to_bf16<<<(unsigned)((nA / 8 + 255) / 256), 256, 0, stream>>>(
      inp, (unsigned int*)Abf, nA);
  cvt_f32_to_bf16<<<(unsigned)((nW / 8 + 255) / 256), 256, 0, stream>>>(
      W, (unsigned int*)Wbf, nW);

  (void)hipFuncSetAttribute((const void*)lif_fused,
                            hipFuncAttributeMaxDynamicSharedMemorySize, SMEM_BYTES);

  dim3 grid(HDIM / TILE_N, BATCH);   // (16, 256)
  lif_fused<<<grid, 256, SMEM_BYTES, stream>>>(Abf, Wbf, bias, spikes, hfinal);
}
